// SelfAttention1d_9569187135754
// MI455X (gfx1250) — compile-verified, matrix operand order checked by reading, not run on hardware
//
#include <hip/hip_runtime.h>
#include <stdint.h>

#define DEVINL __device__ __forceinline__

typedef __attribute__((ext_vector_type(16))) __bf16 v16bf;
typedef __attribute__((ext_vector_type(8)))  __bf16 v8bf;
typedef __attribute__((ext_vector_type(8)))  float  v8f;
typedef __attribute__((ext_vector_type(8)))  unsigned short v8u16;
typedef __attribute__((ext_vector_type(4)))  unsigned int   v4u32;

typedef __attribute__((address_space(3))) unsigned short lds_u16_t;

static constexpr int N_ = 8;
static constexpr int C_ = 1024;
static constexpr int S_ = 2048;
static constexpr int H_ = 8;
static constexpr int D_ = 128;           // C_/H_
static constexpr float EPS_ = 1e-5f;
static constexpr float QK_SCALE = 0.29730177875068026f; // 128^-0.25

// float -> bf16 via native conversion (v_cvt on gfx1250)
DEVINL unsigned short f2bf(float f) {
  __bf16 h = (__bf16)f;
  return __builtin_bit_cast(unsigned short, h);
}

// LDS byte offset of a __shared__ object (addrspace(3) pointers are 32-bit
// offsets from LDS_BASE, which is what GLOBAL_LOAD_ASYNC_TO_LDS wants in VDST).
DEVINL unsigned lds_off(const unsigned short* p) {
  return (unsigned)(uintptr_t)(lds_u16_t*)p;
}

// CDNA5 async copy: global -> LDS, 16B per lane, tracked by ASYNCcnt.
DEVINL void async_load_b128(unsigned ldsoff, const void* gaddr) {
  asm volatile("global_load_async_to_lds_b128 %0, %1, off"
               :
               : "v"(ldsoff), "v"((unsigned long long)(uintptr_t)gaddr)
               : "memory");
}
DEVINL void wait_async0() {
  asm volatile("s_wait_asynccnt 0" ::: "memory");
}

// Build a 16-element bf16 WMMA fragment from two contiguous 8-element chunks.
// Per CDNA5 ISA layout, the two chunks are K {0..7,16..23} for lanes 0-15 and
// K {8..15,24..31} for lanes 16-31 (caller passes the right addresses).
DEVINL v16bf make_frag(const unsigned short* p0, const unsigned short* p1) {
  v8bf a = *reinterpret_cast<const v8bf*>(p0);
  v8bf b = *reinterpret_cast<const v8bf*>(p1);
  v16bf r;
#pragma unroll
  for (int i = 0; i < 8; ++i) { r[i] = a[i]; r[i + 8] = b[i]; }
  return r;
}

DEVINL v8f wmma_bf16(v16bf a, v16bf b, v8f c) {
  return __builtin_amdgcn_wmma_f32_16x16x32_bf16(
      /*neg_a=*/false, a, /*neg_b=*/false, b,
      /*c_mod=*/(short)0, c, /*reuse_a=*/false, /*reuse_b=*/false);
}

// ---------------------------------------------------------------------------
// 1) Per-sample sum / sumsq (mean over axes (C,S))
// ---------------------------------------------------------------------------
__global__ __launch_bounds__(256) void stats_partial_kernel(
    const float* __restrict__ x, float* __restrict__ stats) {
  __shared__ float s0[256];
  __shared__ float s1[256];
  const int n = blockIdx.y;
  const float* xb = x + (size_t)n * C_ * S_;
  const size_t total = (size_t)C_ * S_;
  float a = 0.f, b = 0.f;
  for (size_t i = (size_t)blockIdx.x * blockDim.x + threadIdx.x; i < total;
       i += (size_t)gridDim.x * blockDim.x) {
    float v = xb[i];
    a += v;
    b += v * v;
  }
  s0[threadIdx.x] = a;
  s1[threadIdx.x] = b;
  __syncthreads();
  for (int st = 128; st > 0; st >>= 1) {
    if ((int)threadIdx.x < st) {
      s0[threadIdx.x] += s0[threadIdx.x + st];
      s1[threadIdx.x] += s1[threadIdx.x + st];
    }
    __syncthreads();
  }
  if (threadIdx.x == 0) {
    atomicAdd(&stats[n * 4 + 0], s0[0]);
    atomicAdd(&stats[n * 4 + 1], s1[0]);
  }
}

__global__ void stats_finalize_kernel(float* __restrict__ stats) {
  int n = threadIdx.x;
  if (n < N_) {
    float cnt = (float)((size_t)C_ * S_);
    float mu = stats[n * 4 + 0] / cnt;
    float var = stats[n * 4 + 1] / cnt - mu * mu;
    stats[n * 4 + 2] = mu;
    stats[n * 4 + 3] = rsqrtf(var + EPS_);
  }
}

// ---------------------------------------------------------------------------
// 2) Normalize + affine + transpose:  x[N][C][S] f32 -> xnT[N][S][C] bf16
// ---------------------------------------------------------------------------
__global__ __launch_bounds__(256) void norm_transpose_kernel(
    const float* __restrict__ x, const float* __restrict__ gw,
    const float* __restrict__ gb, const float* __restrict__ stats,
    unsigned short* __restrict__ xnT) {
  __shared__ unsigned short tile[32][33];
  const int n = blockIdx.z;
  const int sB = blockIdx.x * 32;
  const int cB = blockIdx.y * 32;
  const float mu = stats[n * 4 + 2];
  const float rs = stats[n * 4 + 3];
  const int tx = threadIdx.x;  // 0..31
  const int ty = threadIdx.y;  // 0..7
#pragma unroll
  for (int i = 0; i < 4; ++i) {
    int c = cB + ty + i * 8;
    int s = sB + tx;
    float v = x[((size_t)n * C_ + c) * S_ + s];
    float xn = (v - mu) * rs * gw[c] + gb[c];
    tile[ty + i * 8][tx] = f2bf(xn);
  }
  __syncthreads();
#pragma unroll
  for (int i = 0; i < 4; ++i) {
    int s = sB + ty + i * 8;
    int c = cB + tx;
    xnT[((size_t)n * S_ + s) * C_ + c] = tile[tx][ty + i * 8];
  }
}

// ---------------------------------------------------------------------------
// 3) Weight f32 -> bf16
// ---------------------------------------------------------------------------
__global__ void f32_to_bf16_kernel(const float* __restrict__ in,
                                   unsigned short* __restrict__ out, int nels) {
  int i = blockIdx.x * blockDim.x + threadIdx.x;
  if (i < nels) out[i] = f2bf(in[i]);
}

// ---------------------------------------------------------------------------
// 4) QKV GEMM:  qkv[o][s] = sum_c w_qkv[o][c] * xn[c][s]   (per batch n)
// ---------------------------------------------------------------------------
__global__ __launch_bounds__(256) void gemm_qkv_kernel(
    const unsigned short* __restrict__ wq,   // [3C][C] bf16
    const unsigned short* __restrict__ xnT,  // [N][S][C] bf16
    const float* __restrict__ b_qkv,
    unsigned short* __restrict__ qT, unsigned short* __restrict__ kT,
    unsigned short* __restrict__ vv) {
  const int lane = threadIdx.x & 31;
  const int wave = threadIdx.x >> 5;
  const int half = lane >> 4;
  const int lc = lane & 15;
  const int NT = S_ / 64;       // 32
  const int MT = 3 * C_ / 16;   // 192
  int wid = blockIdx.x * 8 + wave;
  int n = wid / (MT * NT);
  int rem = wid % (MT * NT);
  int mt = rem / NT;
  int nt = rem % NT;
  const int o0 = mt * 16;
  const int s0 = nt * 64;

  v8f acc[4] = {};
  const unsigned short* arow = wq + (size_t)(o0 + lc) * C_;
  const unsigned short* bbase = xnT + (size_t)n * S_ * C_;
  for (int k0 = 0; k0 < C_; k0 += 32) {
    __builtin_prefetch(arow + k0 + 128, 0, 1);
    v16bf a = make_frag(arow + k0 + half * 8, arow + k0 + 16 + half * 8);
#pragma unroll
    for (int sub = 0; sub < 4; ++sub) {
      const unsigned short* bcol =
          bbase + (size_t)(s0 + sub * 16 + lc) * C_ + k0;
      v16bf b = make_frag(bcol + half * 8, bcol + 16 + half * 8);
      acc[sub] = wmma_bf16(a, b, acc[sub]);
    }
  }

  const int part = o0 / C_;       // 0=q,1=k,2=v (16-row tile never crosses)
  const int oin = o0 % C_;
  const int h = oin / D_;
  const int dbase = oin % D_;

  if (part < 2) {
    unsigned short* dst = (part == 0) ? qT : kT;
#pragma unroll
    for (int sub = 0; sub < 4; ++sub) {
      int s = s0 + sub * 16 + lc;
      v8u16 pack;
#pragma unroll
      for (int r = 0; r < 8; ++r) {
        int o = o0 + half * 8 + r;
        pack[r] = f2bf((acc[sub][r] + b_qkv[o]) * QK_SCALE);
      }
      *reinterpret_cast<v8u16*>(
          dst + (((size_t)n * H_ + h) * S_ + s) * D_ + dbase + half * 8) = pack;
    }
  } else {
#pragma unroll
    for (int sub = 0; sub < 4; ++sub) {
      int s = s0 + sub * 16 + lc;
#pragma unroll
      for (int r = 0; r < 8; ++r) {
        int o = o0 + half * 8 + r;
        int d = dbase + half * 8 + r;
        vv[(((size_t)n * H_ + h) * D_ + d) * S_ + s] =
            f2bf(acc[sub][r] + b_qkv[o]);
      }
    }
  }
}

// ---------------------------------------------------------------------------
// 5) Flash-style attention. 8 waves/block share (n,h); K/V blocks staged via
//    CDNA5 GLOBAL_LOAD_ASYNC_TO_LDS_B128 (ASYNCcnt); online softmax per wave.
// ---------------------------------------------------------------------------
__global__ __launch_bounds__(256) void attention_kernel(
    const unsigned short* __restrict__ qT, const unsigned short* __restrict__ kT,
    const unsigned short* __restrict__ vv, unsigned short* __restrict__ yT) {
  __shared__ unsigned short Kb[32][128];      // [key][d]
  __shared__ unsigned short Vb[128][32];      // [d][key]
  __shared__ unsigned short Pl[8][16][32];    // per-wave P relayout buffer

  const int tid = threadIdx.x;
  const int lane = tid & 31;
  const int wave = tid >> 5;
  const int half = lane >> 4;
  const int lc = lane & 15;

  const int bid = blockIdx.x;
  const int n = bid / (H_ * 16);
  const int rem = bid % (H_ * 16);
  const int h = rem / 16;
  const int qblk = rem % 16;
  const int q0 = (qblk * 8 + wave) * 16;

  const unsigned short* qbase = qT + ((size_t)n * H_ + h) * S_ * D_;
  const unsigned short* kbase = kT + ((size_t)n * H_ + h) * S_ * D_;
  const unsigned short* vbase = vv + ((size_t)n * H_ + h) * D_ * S_;

  // Per-thread staging coordinates (fixed across iterations).
  const int krow = tid >> 3;          // 0..31  (key)
  const int kcol = (tid & 7) * 16;    // 0..112 (d)
  const int vd = tid >> 1;            // 0..127 (d)
  const int vsc = (tid & 1) * 16;     // 0 | 16 (key)
  const unsigned k_lds0 = lds_off(&Kb[krow][kcol]);
  const unsigned k_lds1 = lds_off(&Kb[krow][kcol + 8]);
  const unsigned v_lds0 = lds_off(&Vb[vd][vsc]);
  const unsigned v_lds1 = lds_off(&Vb[vd][vsc + 8]);

  // Preload Q fragments: 4 chunks of K=32 over the D=128 dimension.
  v16bf qf[4];
  {
    const unsigned short* qrow = qbase + (size_t)(q0 + lc) * D_;
#pragma unroll
    for (int kc = 0; kc < 4; ++kc)
      qf[kc] = make_frag(qrow + kc * 32 + half * 8,
                         qrow + kc * 32 + 16 + half * 8);
  }

  v8f acc[8] = {};
  float rm[8], rl[8];
#pragma unroll
  for (int r = 0; r < 8; ++r) { rm[r] = -1e30f; rl[r] = 0.f; }

  for (int kb = 0; kb < S_ / 32; ++kb) {
    __syncthreads();  // all waves done reading the previous K/V block
    {
      // Async-stage K block (32 keys x 128 d) and V block (128 d x 32 keys):
      // 4 x b128 per thread, no VGPR round-trip, tracked by ASYNCcnt.
      const unsigned short* ksrc = kbase + (size_t)(kb * 32 + krow) * D_ + kcol;
      async_load_b128(k_lds0, ksrc);
      async_load_b128(k_lds1, ksrc + 8);
      const unsigned short* vsrc = vbase + (size_t)vd * S_ + kb * 32 + vsc;
      async_load_b128(v_lds0, vsrc);
      async_load_b128(v_lds1, vsrc + 8);
      wait_async0();
    }
    __syncthreads();  // K/V visible to all waves

    // Scores: two 16x16 tiles (keys kb*32 .. +15, +16 .. +31), K = D = 128.
    v8f s1 = {};
    v8f s2 = {};
#pragma unroll
    for (int kc = 0; kc < 4; ++kc) {
      v16bf b1 = make_frag(&Kb[lc][kc * 32 + half * 8],
                           &Kb[lc][kc * 32 + 16 + half * 8]);
      s1 = wmma_bf16(qf[kc], b1, s1);
    }
#pragma unroll
    for (int kc = 0; kc < 4; ++kc) {
      v16bf b2 = make_frag(&Kb[16 + lc][kc * 32 + half * 8],
                           &Kb[16 + lc][kc * 32 + 16 + half * 8]);
      s2 = wmma_bf16(qf[kc], b2, s2);
    }

    // Online softmax: rows live in 16-lane halves; xor masks 1/2/4/8 stay
    // within a half on wave32.
#pragma unroll
    for (int r = 0; r < 8; ++r) {
      float a = s1[r], b = s2[r];
      float t = fmaxf(a, b);
      t = fmaxf(t, __shfl_xor(t, 1, 32));
      t = fmaxf(t, __shfl_xor(t, 2, 32));
      t = fmaxf(t, __shfl_xor(t, 4, 32));
      t = fmaxf(t, __shfl_xor(t, 8, 32));
      float mnew = fmaxf(rm[r], t);
      float corr = __expf(rm[r] - mnew);
      float p1 = __expf(a - mnew);
      float p2 = __expf(b - mnew);
      float ps = p1 + p2;
      ps += __shfl_xor(ps, 1, 32);
      ps += __shfl_xor(ps, 2, 32);
      ps += __shfl_xor(ps, 4, 32);
      ps += __shfl_xor(ps, 8, 32);
      rl[r] = rl[r] * corr + ps;
      rm[r] = mnew;
      int row = half * 8 + r;
      Pl[wave][row][lc] = f2bf(p1);
      Pl[wave][row][16 + lc] = f2bf(p2);
#pragma unroll
      for (int j = 0; j < 8; ++j) acc[j][r] *= corr;
    }

    // acc += P(16x32) * V(32x128) : 8 d-chunks of 16.
    v16bf pf = make_frag(&Pl[wave][lc][half * 8], &Pl[wave][lc][16 + half * 8]);
#pragma unroll
    for (int j = 0; j < 8; ++j) {
      v16bf bv = make_frag(&Vb[j * 16 + lc][half * 8],
                           &Vb[j * 16 + lc][16 + half * 8]);
      acc[j] = wmma_bf16(pf, bv, acc[j]);
    }
  }

  // Normalize by softmax denominator and store y[n][h][q][d].
  unsigned short* ybase = yT + ((size_t)n * H_ + h) * S_ * D_;
#pragma unroll
  for (int r = 0; r < 8; ++r) {
    float inv = 1.0f / rl[r];
    int row = q0 + half * 8 + r;
#pragma unroll
    for (int j = 0; j < 8; ++j) {
      ybase[(size_t)row * D_ + j * 16 + lc] = f2bf(acc[j][r] * inv);
    }
  }
}

// ---------------------------------------------------------------------------
// 6) Output projection + bias + residual:
//    out[n][oc][s] = x[n][oc][s] + b_out[oc] + sum_c w_out[oc][c]*y[n][c][s]
// ---------------------------------------------------------------------------
__global__ __launch_bounds__(256) void gemm_out_kernel(
    const unsigned short* __restrict__ wo,  // [C][C] bf16
    const unsigned short* __restrict__ yT,  // [N][H][S][D] bf16
    const float* __restrict__ b_out, const float* __restrict__ x,
    float* __restrict__ out) {
  const int lane = threadIdx.x & 31;
  const int wave = threadIdx.x >> 5;
  const int half = lane >> 4;
  const int lc = lane & 15;
  const int NT = S_ / 64;  // 32
  const int MT = C_ / 16;  // 64
  int wid = blockIdx.x * 8 + wave;
  int n = wid / (MT * NT);
  int rem = wid % (MT * NT);
  int mt = rem / NT;
  int nt = rem % NT;
  const int o0 = mt * 16;
  const int s0 = nt * 64;

  v8f acc[4] = {};
  const unsigned short* arow = wo + (size_t)(o0 + lc) * C_;
  for (int k0 = 0; k0 < C_; k0 += 32) {
    __builtin_prefetch(arow + k0 + 128, 0, 1);
    v16bf a = make_frag(arow + k0 + half * 8, arow + k0 + 16 + half * 8);
    const int hk = k0 / D_;        // 32-chunk never crosses a head (32 | 128)
    const int off = k0 % D_;
    const unsigned short* bb = yT + ((size_t)n * H_ + hk) * S_ * D_;
#pragma unroll
    for (int sub = 0; sub < 4; ++sub) {
      const unsigned short* bcol = bb + (size_t)(s0 + sub * 16 + lc) * D_ + off;
      v16bf b = make_frag(bcol + half * 8, bcol + 16 + half * 8);
      acc[sub] = wmma_bf16(a, b, acc[sub]);
    }
  }

#pragma unroll
  for (int sub = 0; sub < 4; ++sub) {
    int s = s0 + sub * 16 + lc;
#pragma unroll
    for (int r = 0; r < 8; ++r) {
      int oc = o0 + half * 8 + r;
      size_t idx = ((size_t)n * C_ + oc) * S_ + s;
      out[idx] = acc[sub][r] + b_out[oc] + x[idx];
    }
  }
}

// ---------------------------------------------------------------------------
extern "C" void kernel_launch(void* const* d_in, const int* in_sizes, int n_in,
                              void* d_out, int out_size, void* d_ws,
                              size_t ws_size, hipStream_t stream) {
  (void)in_sizes; (void)n_in; (void)out_size; (void)ws_size;
  const float* x = (const float*)d_in[0];
  const float* gn_w = (const float*)d_in[1];
  const float* gn_b = (const float*)d_in[2];
  const float* w_qkv = (const float*)d_in[3];
  const float* b_qkv = (const float*)d_in[4];
  const float* w_out = (const float*)d_in[5];
  const float* b_out = (const float*)d_in[6];
  float* out = (float*)d_out;

  char* ws = (char*)d_ws;
  size_t off = 0;
  auto alloc = [&](size_t bytes) -> void* {
    void* p = ws + off;
    off += (bytes + 255) & ~(size_t)255;
    return p;
  };
  float* stats = (float*)alloc(N_ * 4 * sizeof(float));
  unsigned short* wq_bf = (unsigned short*)alloc((size_t)3 * C_ * C_ * 2);
  unsigned short* wo_bf = (unsigned short*)alloc((size_t)C_ * C_ * 2);
  unsigned short* xnT = (unsigned short*)alloc((size_t)N_ * S_ * C_ * 2);
  unsigned short* qT = (unsigned short*)alloc((size_t)N_ * C_ * S_ * 2);
  unsigned short* kT = (unsigned short*)alloc((size_t)N_ * C_ * S_ * 2);
  unsigned short* vv = (unsigned short*)alloc((size_t)N_ * C_ * S_ * 2);
  unsigned short* yT = xnT;  // xnT is dead after the QKV GEMM -> alias

  hipMemsetAsync(stats, 0, N_ * 4 * sizeof(float), stream);
  stats_partial_kernel<<<dim3(64, N_), 256, 0, stream>>>(x, stats);
  stats_finalize_kernel<<<1, 32, 0, stream>>>(stats);

  {
    int ne = 3 * C_ * C_;
    f32_to_bf16_kernel<<<(ne + 255) / 256, 256, 0, stream>>>(w_qkv, wq_bf, ne);
    ne = C_ * C_;
    f32_to_bf16_kernel<<<(ne + 255) / 256, 256, 0, stream>>>(w_out, wo_bf, ne);
  }

  norm_transpose_kernel<<<dim3(S_ / 32, C_ / 32, N_), dim3(32, 8), 0, stream>>>(
      x, gn_w, gn_b, stats, xnT);

  {
    int waves = N_ * (3 * C_ / 16) * (S_ / 64);  // 49152
    gemm_qkv_kernel<<<waves / 8, 256, 0, stream>>>(wq_bf, xnT, b_qkv, qT, kT,
                                                   vv);
  }

  attention_kernel<<<N_ * H_ * 16, 256, 0, stream>>>(qT, kT, vv, yT);

  {
    int waves = N_ * (C_ / 16) * (S_ / 64);  // 16384
    gemm_out_kernel<<<waves / 8, 256, 0, stream>>>(wo_bf, yT, b_out, x, out);
  }
}
